// SMPL_89472758711055
// MI455X (gfx1250) — compile-verified, matrix operand order checked by reading, not run on hardware
//
#include <hip/hip_runtime.h>
#include <hip/hip_bf16.h>

typedef __attribute__((ext_vector_type(16))) _Float16 v16h;
typedef __attribute__((ext_vector_type(8)))  _Float16 v8h;
typedef __attribute__((ext_vector_type(8)))  float    v8f;
typedef __attribute__((ext_vector_type(2)))  float    v2f;

#define B_    1024
#define V_    6890
#define J_    24
#define NB_   10
#define KDIM  207            // (J-1)*9
#define KPAD  224            // 7 * 32
#define NCOLS (V_ * 3)       // 20670
#define NPAD  20672          // 1292 * 16  (also 646*32)
#define MTILES 64            // 1024 / 16
#define NTILES 1292
#define VT_   431            // ceil(V/16) vertex tiles

// ---------------- workspace layout (bytes) ----------------
static const size_t OFF_VP   = 0;                                   // f32 [B][NCOLS]   (v_shaped, then in-place v_posed)
static const size_t OFF_J    = OFF_VP   + (size_t)B_ * NCOLS * 4;   // f32 [B][J][3]
static const size_t OFF_ROT  = OFF_J    + (size_t)B_ * J_ * 3 * 4;  // f32 [B][J][9]
static const size_t OFF_AF   = OFF_ROT  + (size_t)B_ * J_ * 9 * 4;  // f16 [B][KPAD]    pose_feat
static const size_t OFF_BT   = OFF_AF   + (size_t)B_ * KPAD * 2;    // f16 [NPAD][KPAD] posedirs^T
static const size_t OFF_AREL = OFF_BT   + (size_t)NPAD * KPAD * 2;  // f32 [B][J][16]   A_rel padded to 16 cols

// ---------------- K1: shape blendshapes ----------------
__global__ void smpl_shape(const float* __restrict__ shape, const float* __restrict__ vt,
                           const float* __restrict__ sd, float* __restrict__ vsh) {
  size_t idx = (size_t)blockIdx.x * blockDim.x + threadIdx.x;
  if (idx >= (size_t)B_ * V_) return;
  int b = (int)(idx / V_);
  int v = (int)(idx % V_);
  float sh[NB_];
#pragma unroll
  for (int l = 0; l < NB_; ++l) sh[l] = shape[b * NB_ + l];
#pragma unroll
  for (int c = 0; c < 3; ++c) {
    const float* sdp = sd + (size_t)(v * 3 + c) * NB_;
    float acc = vt[v * 3 + c];
#pragma unroll
    for (int l = 0; l < NB_; ++l) acc += sh[l] * sdp[l];
    vsh[(size_t)b * NCOLS + v * 3 + c] = acc;
  }
}

// ---------------- K2: joint regression (block reduction over V) ----------------
__global__ void smpl_joints(const float* __restrict__ jreg, const float* __restrict__ vsh,
                            float* __restrict__ joints) {
  int b = blockIdx.x / J_;
  int j = blockIdx.x % J_;
  __shared__ float red[3][256];
  float s0 = 0.f, s1 = 0.f, s2 = 0.f;
  for (int v = threadIdx.x; v < V_; v += 256) {
    float w = jreg[(size_t)j * V_ + v];
    const float* p = vsh + (size_t)b * NCOLS + v * 3;
    s0 += w * p[0]; s1 += w * p[1]; s2 += w * p[2];
  }
  red[0][threadIdx.x] = s0; red[1][threadIdx.x] = s1; red[2][threadIdx.x] = s2;
  __syncthreads();
  for (int off = 128; off > 0; off >>= 1) {
    if ((int)threadIdx.x < off) {
      red[0][threadIdx.x] += red[0][threadIdx.x + off];
      red[1][threadIdx.x] += red[1][threadIdx.x + off];
      red[2][threadIdx.x] += red[2][threadIdx.x + off];
    }
    __syncthreads();
  }
  if (threadIdx.x == 0) {
    float* o = joints + (size_t)(b * J_ + j) * 3;
    o[0] = red[0][0]; o[1] = red[1][0]; o[2] = red[2][0];
  }
}

// ---------------- K3: rodrigues + f16 pose features ----------------
__global__ void smpl_rodrigues(const float* __restrict__ pelvis, const float* __restrict__ body,
                               float* __restrict__ rot, _Float16* __restrict__ Af) {
  int idx = blockIdx.x * blockDim.x + threadIdx.x;
  if (idx >= B_ * J_) return;
  int b = idx / J_, j = idx % J_;
  const float* r = (j == 0) ? (pelvis + (size_t)b * 3)
                            : (body + ((size_t)b * (J_ - 1) + (j - 1)) * 3);
  float x = r[0], y = r[1], z = r[2];
  float th = sqrtf(x * x + y * y + z * z + 1e-8f);
  float inv = 1.f / th;
  float ax = x * inv, ay = y * inv, az = z * inv;
  float c = cosf(th), s = sinf(th), ic = 1.f - c;
  float R[9];
  R[0] = c + ic * ax * ax;      R[1] = ic * ax * ay - s * az; R[2] = ic * ax * az + s * ay;
  R[3] = ic * ay * ax + s * az; R[4] = c + ic * ay * ay;      R[5] = ic * ay * az - s * ax;
  R[6] = ic * az * ax - s * ay; R[7] = ic * az * ay + s * ax; R[8] = c + ic * az * az;
  float* rp = rot + (size_t)(b * J_ + j) * 9;
#pragma unroll
  for (int e = 0; e < 9; ++e) rp[e] = R[e];
  if (j > 0) {
    _Float16* ap = Af + (size_t)b * KPAD + (j - 1) * 9;
#pragma unroll
    for (int e = 0; e < 9; ++e)
      ap[e] = (_Float16)(R[e] - ((e == 0 || e == 4 || e == 8) ? 1.f : 0.f));
  } else {
    _Float16* ap = Af + (size_t)b * KPAD + KDIM;
#pragma unroll
    for (int i = 0; i < KPAD - KDIM; ++i) ap[i] = (_Float16)0.f;
  }
}

// ---------------- K4: posedirs (207x20670 f32) -> transposed padded f16 [NPAD][KPAD] ----------------
__global__ void smpl_cvt_posedirs(const float* __restrict__ pd, _Float16* __restrict__ Bt) {
  __shared__ float tile[32][33];
  int n0 = blockIdx.x * 32;
  int k0 = blockIdx.y * 32;
#pragma unroll
  for (int i = 0; i < 4; ++i) {
    int k = k0 + threadIdx.y + i * 8;
    int n = n0 + threadIdx.x;
    float v = 0.f;
    if (k < KDIM && n < NCOLS) v = pd[(size_t)k * NCOLS + n];
    tile[threadIdx.y + i * 8][threadIdx.x] = v;
  }
  __syncthreads();
#pragma unroll
  for (int i = 0; i < 4; ++i) {
    int n = n0 + threadIdx.y + i * 8;   // < NPAD always (646*32)
    int k = k0 + threadIdx.x;           // < KPAD always (7*32)
    Bt[(size_t)n * KPAD + k] = (_Float16)tile[threadIdx.x][threadIdx.y + i * 8];
  }
}

// ---------------- K5: pose GEMM via v_wmma_f32_16x16x32_f16, in-place v_posed add ----------------
__global__ void smpl_pose_gemm(const _Float16* __restrict__ Af,
                               const _Float16* __restrict__ Bt,
                               float* __restrict__ vposed) {
  int wave = threadIdx.x >> 5;
  int lane = threadIdx.x & 31;
  int tile = blockIdx.x * 8 + wave;           // wave-uniform
  if (tile >= MTILES * NTILES) return;        // uniform exit, EXEC stays all-ones
  int tm = tile % MTILES;                     // M fast -> 8 waves/block share the B tile
  int tn = tile / MTILES;
  int half = lane >> 4;
  int l15  = lane & 15;
  const _Float16* Arow = Af + (size_t)(tm * 16 + l15) * KPAD;  // A: M = l15 in both halves
  const _Float16* Brow = Bt + (size_t)(tn * 16 + l15) * KPAD;  // B^T row = column N of B
  v8f acc = {};
#pragma unroll
  for (int kt = 0; kt < KPAD; kt += 32) {
    // ISA 16-bit A/B layout: lanes 0-15 hold K {kt..kt+7, kt+16..kt+23}, lanes 16-31 offset +8
    v8h alo = *(const v8h*)(Arow + kt + half * 8);
    v8h ahi = *(const v8h*)(Arow + kt + 16 + half * 8);
    v8h blo = *(const v8h*)(Brow + kt + half * 8);
    v8h bhi = *(const v8h*)(Brow + kt + 16 + half * 8);
    v16h a, b;
#pragma unroll
    for (int i = 0; i < 8; ++i) {
      a[i] = alo[i]; a[8 + i] = ahi[i];
      b[i] = blo[i]; b[8 + i] = bhi[i];
    }
    acc = __builtin_amdgcn_wmma_f32_16x16x32_f16(
        /*neg_a=*/false, a, /*neg_b=*/false, b,
        /*c_mod=*/(short)0, acc, /*reuse_a=*/false, /*reuse_b=*/false);
  }
  int col = tn * 16 + l15;
  if (col < NCOLS) {
#pragma unroll
    for (int r = 0; r < 8; ++r) {           // D: VGPR r -> M = r + 8*half, N = l15
      int m = tm * 16 + half * 8 + r;
      float* p = vposed + (size_t)m * NCOLS + col;
      *p = *p + acc[r];
    }
  }
}

// ---------------- K6: forward kinematics (one thread per batch, fully unrolled chain) ----------------
// Writes A_rel padded to 16 columns: arel16[b][j][0..11] = 3x4 transform, [12..15] = 0
__global__ void smpl_fk(const float* __restrict__ rot, const float* __restrict__ joints,
                        float* __restrict__ arel16) {
  int b = blockIdx.x * blockDim.x + threadIdx.x;
  if (b >= B_) return;
  const int PAR[J_] = {-1,0,0,0,1,2,3,4,5,6,7,8,9,9,9,12,13,14,16,17,18,19,20,21};
  float C[J_][12];   // chain of 3x4 affine transforms, constant-indexed (register resident)
#pragma unroll
  for (int j = 0; j < J_; ++j) {
    float R[9];
#pragma unroll
    for (int e = 0; e < 9; ++e) R[e] = rot[(size_t)(b * J_ + j) * 9 + e];
    const float* jj = joints + (size_t)(b * J_ + j) * 3;
    float jx = jj[0], jy = jj[1], jz = jj[2];
    float tx = jx, ty = jy, tz = jz;
    if (j > 0) {
      const float* jp = joints + (size_t)(b * J_ + PAR[j]) * 3;
      tx -= jp[0]; ty -= jp[1]; tz -= jp[2];
    }
    if (j == 0) {
      C[0][0] = R[0]; C[0][1] = R[1]; C[0][2]  = R[2]; C[0][3]  = tx;
      C[0][4] = R[3]; C[0][5] = R[4]; C[0][6]  = R[5]; C[0][7]  = ty;
      C[0][8] = R[6]; C[0][9] = R[7]; C[0][10] = R[8]; C[0][11] = tz;
    } else {
      const int p = PAR[j];
#pragma unroll
      for (int row = 0; row < 3; ++row) {
        float g0 = C[p][row * 4 + 0], g1 = C[p][row * 4 + 1], g2 = C[p][row * 4 + 2], g3 = C[p][row * 4 + 3];
        C[j][row * 4 + 0] = g0 * R[0] + g1 * R[3] + g2 * R[6];
        C[j][row * 4 + 1] = g0 * R[1] + g1 * R[4] + g2 * R[7];
        C[j][row * 4 + 2] = g0 * R[2] + g1 * R[5] + g2 * R[8];
        C[j][row * 4 + 3] = g0 * tx   + g1 * ty   + g2 * tz   + g3;
      }
    }
    // A_rel: translation corrected by rest-pose joint location; pad cols 12..15 with zero
    float* o = arel16 + (size_t)(b * J_ + j) * 16;
#pragma unroll
    for (int row = 0; row < 3; ++row) {
      float g0 = C[j][row * 4 + 0], g1 = C[j][row * 4 + 1], g2 = C[j][row * 4 + 2];
      o[row * 4 + 0] = g0;
      o[row * 4 + 1] = g1;
      o[row * 4 + 2] = g2;
      o[row * 4 + 3] = C[j][row * 4 + 3] - (g0 * jx + g1 * jy + g2 * jz);
    }
    o[12] = 0.f; o[13] = 0.f; o[14] = 0.f; o[15] = 0.f;
  }
}

// ---------------- K7: LBS via v_wmma_f32_16x16x4_f32 (full fp32), LDS-staged D tile ----------------
// Per wave: one 16-vertex tile of one batch. D = W_tile(16xK24) x A_rel(24x16) in 6 K4 steps.
__global__ void smpl_lbs_wmma(const float* __restrict__ W, const float* __restrict__ arel16,
                              const float* __restrict__ vposed, const float* __restrict__ trans,
                              float* __restrict__ out) {
  __shared__ float tv[8][16][17];
  int wave = threadIdx.x >> 5;
  int lane = threadIdx.x & 31;
  int tile = blockIdx.x * 8 + wave;   // grid is exact: B_*VT_ = 441344 = 55168*8
  int b  = tile / VT_;
  int vt = tile % VT_;
  int v0 = vt * 16;
  int half = lane >> 4;
  int l15  = lane & 15;
  int m = v0 + l15;                   // vertex row for A fragment
  if (m >= V_) m = V_ - 1;            // clamp padded rows (no EXEC divergence)
  const float* wrow = W + (size_t)m * J_;
  const float* bcol = arel16 + (size_t)b * J_ * 16 + l15;   // column N = l15 of A_rel
  v8f acc = {};
#pragma unroll
  for (int kt = 0; kt < J_; kt += 4) {
    // f32 A 16x4: lanes 0-15 K={kt,kt+1}, lanes 16-31 K={kt+2,kt+3}; B mirrored per half
    int k = kt + half * 2;
    v2f a = *(const v2f*)(wrow + k);                 // 8-byte load, 8B aligned (row = 96B)
    v2f bb;
    bb[0] = bcol[(size_t)k * 16];
    bb[1] = bcol[(size_t)(k + 1) * 16];
    acc = __builtin_amdgcn_wmma_f32_16x16x4_f32(
        /*neg_a=*/false, a, /*neg_b=*/false, bb,
        /*c_mod=*/(short)0, acc, /*reuse_a=*/false, /*reuse_b=*/false);
  }
  // D layout: VGPR r -> vertex row (half*8 + r), Tv element N = l15 ; stage through LDS
#pragma unroll
  for (int r = 0; r < 8; ++r) tv[wave][half * 8 + r][l15] = acc[r];
  __syncthreads();
  if (lane < 16) {
    int v = v0 + lane;
    if (v < V_) {
      const float* T = tv[wave][lane];
      const float* p = vposed + (size_t)b * NCOLS + v * 3;
      float p0 = p[0], p1 = p[1], p2 = p[2];
      float x = T[0] * p0 + T[1] * p1 + T[2]  * p2 + T[3];
      float y = T[4] * p0 + T[5] * p1 + T[6]  * p2 + T[7];
      float z = T[8] * p0 + T[9] * p1 + T[10] * p2 + T[11];
      size_t o = ((size_t)b * V_ + v) * 3;
      out[o + 0] = x + trans[b * 3 + 0];
      out[o + 1] = y + trans[b * 3 + 1];
      out[o + 2] = z + trans[b * 3 + 2];
    }
  }
}

extern "C" void kernel_launch(void* const* d_in, const int* in_sizes, int n_in,
                              void* d_out, int out_size, void* d_ws, size_t ws_size,
                              hipStream_t stream) {
  const float* shape     = (const float*)d_in[0];
  const float* body_pose = (const float*)d_in[1];
  const float* pelvis    = (const float*)d_in[2];
  const float* trans     = (const float*)d_in[3];
  const float* v_temp    = (const float*)d_in[4];
  const float* shapedirs = (const float*)d_in[5];
  const float* posedirs  = (const float*)d_in[6];
  const float* lbsw      = (const float*)d_in[7];
  const float* jreg      = (const float*)d_in[8];
  // d_in[9] = parents: baked in at compile time (standard SMPL tree)

  char* ws = (char*)d_ws;
  float*    vposed = (float*)(ws + OFF_VP);
  float*    joints = (float*)(ws + OFF_J);
  float*    rot    = (float*)(ws + OFF_ROT);
  _Float16* Af     = (_Float16*)(ws + OFF_AF);
  _Float16* Bt     = (_Float16*)(ws + OFF_BT);
  float*    arel16 = (float*)(ws + OFF_AREL);
  float*    out    = (float*)d_out;

  const int nBV = (int)(((size_t)B_ * V_ + 255) / 256);

  smpl_shape<<<nBV, 256, 0, stream>>>(shape, v_temp, shapedirs, vposed);
  smpl_joints<<<B_ * J_, 256, 0, stream>>>(jreg, vposed, joints);
  smpl_rodrigues<<<(B_ * J_ + 255) / 256, 256, 0, stream>>>(pelvis, body_pose, rot, Af);
  smpl_cvt_posedirs<<<dim3(NPAD / 32, KPAD / 32), dim3(32, 8), 0, stream>>>(posedirs, Bt);
  smpl_pose_gemm<<<(MTILES * NTILES) / 8, 256, 0, stream>>>(Af, Bt, vposed);
  smpl_fk<<<(B_ + 255) / 256, 256, 0, stream>>>(rot, joints, arel16);
  smpl_lbs_wmma<<<(B_ * VT_) / 8, 256, 0, stream>>>(lbsw, arel16, vposed, trans, out);
}
